// DeepRL_3D_model_65137474011941
// MI455X (gfx1250) — compile-verified, matrix operand order checked by reading, not run on hardware
//
#include <hip/hip_runtime.h>
#include <hip/hip_bf16.h>

typedef __attribute__((ext_vector_type(16))) __bf16       v16bf;
typedef __attribute__((ext_vector_type(8)))  float        v8f;
typedef __attribute__((ext_vector_type(4)))  unsigned int v4u;
typedef __attribute__((ext_vector_type(8)))  int          v8i;
typedef __attribute__((ext_vector_type(4)))  int          v4i;

#define H 128
#define WAVES 4

__device__ __forceinline__ float silu_f(float x) {
    return x * (1.0f / (1.0f + __expf(-x)));
}

// WMMA 16x16x32 bf16 fragment map (ISA 7.12.2): element i (0..15), half = lane>>4
// -> K offset inside a 32-wide K-step.
__device__ __forceinline__ int kmap(int i, int half) {
    int v = i >> 1;
    return ((v & 4) << 2) + (half << 3) + ((v & 3) << 1) + (i & 1);
}

// Inverse map: natural (row, k) -> linear index in fragment-ordered buffer
// laid out as [k/32][lane][16] (32B per lane per k-step => b128-pair loads).
__device__ __forceinline__ int fragpos(int row, int k) {
    int kt   = k >> 5;
    int kk   = k & 31;
    int half = (kk >> 3) & 1;
    int v    = ((kk & 16) >> 2) + ((kk & 7) >> 1);
    int i    = (v << 1) + (kk & 1);
    return ((kt * 32) + (half << 4) + row) * 16 + i;
}

__device__ __forceinline__ v8f wmma_bf16(v16bf a, v16bf b, v8f c) {
    return __builtin_amdgcn_wmma_f32_16x16x32_bf16(false, a, false, b, (short)0, c, false, false);
}

// ---------------------------------------------------------------- prep kernels
// Swizzle f32 weight [K][N] (leading dim srcLd, live region kLive x nLive, zero pad)
// into bf16 WMMA-fragment order: [N/16][K/32][32 lanes][16 elems].
__global__ void swz_kernel(const float* __restrict__ src, __bf16* __restrict__ dst,
                           int K, int N, int kLive, int nLive, int srcLd) {
    int i = blockIdx.x * blockDim.x + threadIdx.x;
    if (i >= K * N) return;
    int e    = i & 15;
    int lane = (i >> 4) & 31;
    int kt   = (i >> 9) % (K >> 5);
    int ct   = (i >> 9) / (K >> 5);
    int kk   = (kt << 5) + kmap(e, lane >> 4);
    int n    = (ct << 4) + (lane & 15);
    float v  = (kk < kLive && n < nLive) ? src[(long)kk * srcLd + n] : 0.0f;
    dst[i] = (__bf16)v;
}

__global__ void embed_type_kernel(const float* __restrict__ W, const float* __restrict__ b,
                                  float* __restrict__ d) {
    int i = blockIdx.x * blockDim.x + threadIdx.x;
    if (i < 16 * H) d[i] = W[i] + b[i & (H - 1)];
}

__global__ void zero_kernel(float* __restrict__ p, long n) {
    long i = (long)blockIdx.x * blockDim.x + threadIdx.x;
    if (i < n) p[i] = 0.0f;
}

__global__ void copy_kernel(const float* __restrict__ s, float* __restrict__ d, long n) {
    long i = (long)blockIdx.x * blockDim.x + threadIdx.x;
    if (i < n) d[i] = s[i];
}

// node embedding: h = x @ W + b (K = 16 or 27), plus bf16 shadow copy
__global__ void embed_kernel(const float* __restrict__ xin, const float* __restrict__ W,
                             const float* __restrict__ b, float* __restrict__ hf,
                             __bf16* __restrict__ hb, int N, int F) {
    int n = blockIdx.x;
    int t = threadIdx.x;
    if (n >= N) return;
    float acc = b[t];
    for (int k = 0; k < F; ++k) acc = fmaf(xin[(long)n * F + k], W[k * H + t], acc);
    hf[(long)n * H + t] = acc;
    hb[(long)n * H + t] = (__bf16)acc;
}

// ---------------------------------------------------------------- EGCL edge kernel
struct __align__(64) EdgeSh {
    __bf16 eA[8 * 32 * 16];   // e_in fragments (K=256), 8KB
    __bf16 mT[4 * 32 * 16];   // MLP-1 output fragments (K=128), 4KB
    float  rel[16][3];
    float  d2[16];
    float  coef[16];
    int    dst[16];
};

__global__ void __launch_bounds__(128)
egcl_edge_kernel(const int* __restrict__ eidx, int E,
                 const float* __restrict__ xpos,
                 const __bf16* __restrict__ hb,
                 const __bf16* __restrict__ W1s,   // frag-swizzled [8][8][32][16]
                 const float* __restrict__ be1g,
                 const float* __restrict__ w1last, // f32 K-row 256 of We1 (d2 column)
                 const __bf16* __restrict__ W2s,   // frag-swizzled [8][4][32][16]
                 const float* __restrict__ be2g,
                 const float* __restrict__ Wxg,
                 const float* __restrict__ bxp,
                 float* __restrict__ agg_m, float* __restrict__ agg_x,
                 float* __restrict__ deg) {
    __shared__ EdgeSh sh[WAVES];
    __shared__ __align__(64) __bf16 w2lds[8 * 4 * 32 * 16];   // 32KB, block-shared We2
    const int lane = threadIdx.x & 31;
    const int wid  = threadIdx.x >> 5;
    EdgeSh& s = sh[wid];
    const int e0 = (blockIdx.x * WAVES + wid) << 4;
    const int* __restrict__ srcA = eidx;
    const int* __restrict__ dstA = eidx + E;

    // ---- stage We2 fragments into LDS via Tensor Data Mover (TDM) ----
#if __has_builtin(__builtin_amdgcn_tensor_load_to_lds)
    if (wid == 0) {
        unsigned lds_off = (unsigned)(uintptr_t)(void*)w2lds;     // LDS aperture: low 32b
        unsigned long long ga = (unsigned long long)(uintptr_t)W2s;
        v4u g0;
        g0[0] = 1u;                                               // count=1
        g0[1] = lds_off;                                          // lds_addr
        g0[2] = (unsigned)(ga & 0xFFFFFFFFu);                     // global_addr[31:0]
        g0[3] = (unsigned)((ga >> 32) & 0x01FFFFFFu) | (2u << 30);// addr[56:32] | type=2
        v8i g1;
        g1[0] = 1 << 16;                 // data_size=2B (workgroup_mask=0)
        g1[1] = (int)(16384u << 16);     // tensor_dim0[15:0]=16384 elems
        g1[2] = 1 << 16;                 // tensor_dim0 hi=0 | tensor_dim1=1
        g1[3] = (int)(16384u << 16);     // tile_dim0=16384 (1-D tile)
        g1[4] = 0;                       // tile_dim1=0 (unused), tile_dim2=0
        g1[5] = 16384;                   // tensor_dim0_stride
        g1[6] = 0;
        g1[7] = 0;
        v4i z = {0, 0, 0, 0};
#if __has_include(<hip/amd_detail/amd_gfx1250_TDM.h>)
        v8i z8 = {0, 0, 0, 0, 0, 0, 0, 0};
        __builtin_amdgcn_tensor_load_to_lds(g0, g1, z, z, z8, 0); // clang-23 6-arg form
#else
        __builtin_amdgcn_tensor_load_to_lds(g0, g1, z, z, 0);     // ROCm7.2 5-arg form
#endif
    }
    __builtin_amdgcn_s_wait_tensorcnt(0);
#else
    {   // fallback: cooperative b128 copy
        const uint4* __restrict__ sv = (const uint4*)W2s;
        uint4* dv = (uint4*)w2lds;
        for (int i = threadIdx.x; i < 2048; i += 128) dv[i] = sv[i];
    }
#endif

    if (lane < 16) {
        int e = e0 + lane;
        int d = -1;
        float dx = 0.f, dy = 0.f, dz = 0.f, d2 = 0.f;
        if (e < E) {
            int si = srcA[e];
            d = dstA[e];
            dx = xpos[d * 3 + 0] - xpos[si * 3 + 0];
            dy = xpos[d * 3 + 1] - xpos[si * 3 + 1];
            dz = xpos[d * 3 + 2] - xpos[si * 3 + 2];
            d2 = dx * dx + dy * dy + dz * dz;
        }
        s.dst[lane] = d;
        s.rel[lane][0] = dx; s.rel[lane][1] = dy; s.rel[lane][2] = dz;
        s.d2[lane] = d2;
        s.coef[lane] = 0.0f;
    }
    // gather [h_dst | h_src] -> fragment-ordered LDS (coalesced global reads)
    for (int idx = lane; idx < 16 * 256; idx += 32) {
        int row = idx >> 8, c = idx & 255;
        int e = e0 + row;
        __bf16 v = (__bf16)0.0f;
        if (e < E) {
            int node = (c < H) ? dstA[e] : srcA[e];
            v = hb[(long)node * H + (c & (H - 1))];
        }
        s.eA[fragpos(row, c)] = v;
    }
    __syncthreads();

    const int mcol = lane & 15;
    const int half = lane >> 4;
    const v16bf* __restrict__ av1 = (const v16bf*)s.eA;
    const v16bf* __restrict__ av2 = (const v16bf*)s.mT;
    const v16bf* __restrict__ bv1 = (const v16bf*)W1s;
    const v16bf* __restrict__ bv2 = (const v16bf*)w2lds;

    // MLP-1: m = silu(e_in @ We1 + be1), K=256 via WMMA + f32 d2 rank-1 term
    for (int ct = 0; ct < 8; ++ct) {
        int n = (ct << 4) + mcol;
        float bcol = be1g[n];
        v8f acc;
#pragma unroll
        for (int r = 0; r < 8; ++r) acc[r] = bcol;
#pragma unroll
        for (int kt = 0; kt < 8; ++kt) {
            v16bf a = av1[(kt << 5) + lane];
            v16bf b = bv1[(((ct << 3) + kt) << 5) + lane];
            acc = wmma_bf16(a, b, acc);
        }
        float wl = w1last[n];
#pragma unroll
        for (int r = 0; r < 8; ++r) {
            int row = r + (half << 3);
            float v = silu_f(acc[r] + s.d2[row] * wl);
            s.mT[fragpos(row, n)] = (__bf16)v;
        }
    }
    __syncthreads();

    // MLP-2: m = silu(m @ We2 + be2); scatter agg_m, reduce coef = m.Wx
    const float bxv = bxp[0];
    for (int ct = 0; ct < 8; ++ct) {
        int n = (ct << 4) + mcol;
        float bcol = be2g[n];
        v8f acc;
#pragma unroll
        for (int r = 0; r < 8; ++r) acc[r] = bcol;
#pragma unroll
        for (int kt = 0; kt < 4; ++kt) {
            v16bf a = av2[(kt << 5) + lane];
            v16bf b = bv2[(((ct << 2) + kt) << 5) + lane];
            acc = wmma_bf16(a, b, acc);
        }
        float wxn = Wxg[n];
#pragma unroll
        for (int r = 0; r < 8; ++r) {
            int row = r + (half << 3);
            float v = silu_f(acc[r]);
            int d = s.dst[row];
            if (d >= 0) atomicAdd(&agg_m[(long)d * H + n], v);
            atomicAdd(&s.coef[row], v * wxn);
        }
    }
    __syncthreads();

    if (lane < 16) {
        int d = s.dst[lane];
        if (d >= 0) {
            float c = s.coef[lane] + bxv;
            atomicAdd(&agg_x[d * 3 + 0], s.rel[lane][0] * c);
            atomicAdd(&agg_x[d * 3 + 1], s.rel[lane][1] * c);
            atomicAdd(&agg_x[d * 3 + 2], s.rel[lane][2] * c);
            atomicAdd(&deg[d], 1.0f);
        }
    }
}

// ---------------------------------------------------------------- EGCL node kernel
struct __align__(64) NodeSh {
    __bf16 A[8 * 32 * 16];      // [h | agg_m] fragments, K=256
    __bf16 hid[4 * 32 * 16];    // hidden fragments, K=128
};

__global__ void __launch_bounds__(128)
egcl_node_kernel(int N, const float* __restrict__ agg_m, const float* __restrict__ agg_x,
                 const float* __restrict__ deg,
                 const __bf16* __restrict__ W1s, const float* __restrict__ bh1g,
                 const __bf16* __restrict__ W2s, const float* __restrict__ bh2g,
                 float* __restrict__ hf, __bf16* __restrict__ hb, float* __restrict__ xpos) {
    __shared__ NodeSh sh[WAVES];
    const int lane = threadIdx.x & 31;
    const int wid  = threadIdx.x >> 5;
    NodeSh& s = sh[wid];
    const int n0 = (blockIdx.x * WAVES + wid) << 4;

    for (int idx = lane; idx < 16 * 256; idx += 32) {
        int row = idx >> 8, c = idx & 255;
        int node = n0 + row;
        __bf16 v = (__bf16)0.0f;
        if (node < N)
            v = (c < H) ? hb[(long)node * H + c] : (__bf16)agg_m[(long)node * H + (c - H)];
        s.A[fragpos(row, c)] = v;
    }
    __syncthreads();

    const int mcol = lane & 15;
    const int half = lane >> 4;
    const v16bf* __restrict__ av1 = (const v16bf*)s.A;
    const v16bf* __restrict__ av2 = (const v16bf*)s.hid;
    const v16bf* __restrict__ bv1 = (const v16bf*)W1s;
    const v16bf* __restrict__ bv2 = (const v16bf*)W2s;

    for (int ct = 0; ct < 8; ++ct) {
        int n = (ct << 4) + mcol;
        float bcol = bh1g[n];
        v8f acc;
#pragma unroll
        for (int r = 0; r < 8; ++r) acc[r] = bcol;
#pragma unroll
        for (int kt = 0; kt < 8; ++kt) {
            v16bf a = av1[(kt << 5) + lane];
            v16bf b = bv1[(((ct << 3) + kt) << 5) + lane];
            acc = wmma_bf16(a, b, acc);
        }
#pragma unroll
        for (int r = 0; r < 8; ++r) {
            int row = r + (half << 3);
            s.hid[fragpos(row, n)] = (__bf16)silu_f(acc[r]);
        }
    }
    __syncthreads();

    for (int ct = 0; ct < 8; ++ct) {
        int n = (ct << 4) + mcol;
        float bcol = bh2g[n];
        v8f acc;
#pragma unroll
        for (int r = 0; r < 8; ++r) acc[r] = bcol;
#pragma unroll
        for (int kt = 0; kt < 4; ++kt) {
            v16bf a = av2[(kt << 5) + lane];
            v16bf b = bv2[(((ct << 2) + kt) << 5) + lane];
            acc = wmma_bf16(a, b, acc);
        }
#pragma unroll
        for (int r = 0; r < 8; ++r) {
            int row = r + (half << 3);
            int node = n0 + row;
            if (node < N) {
                float hnew = hf[(long)node * H + n] + acc[r];
                hf[(long)node * H + n] = hnew;
                hb[(long)node * H + n] = (__bf16)hnew;
            }
        }
    }

    if (lane < 16) {
        int node = n0 + lane;
        if (node < N) {
            float dg = fmaxf(deg[node], 1.0f);
#pragma unroll
            for (int k = 0; k < 3; ++k)
                xpos[node * 3 + k] += agg_x[node * 3 + k] / dg;
        }
    }
}

// ---------------------------------------------------------------- NextType head
struct __align__(64) HeadSh {
    __bf16 A[4 * 32 * 16];    // embed_type * h[n] fragments, K=128
    __bf16 t1[3 * 32 * 16];   // dense1 output fragments, K=96
    float  t2[16][48];
    float  lg[16];
};

__global__ void __launch_bounds__(128)
head_kernel(int NL, const float* __restrict__ hf, const float* __restrict__ embed,
            const __bf16* __restrict__ Wd1s, const float* __restrict__ bd1,
            const __bf16* __restrict__ Wd2s, const float* __restrict__ bd2,
            const float* __restrict__ Wd3, const float* __restrict__ bd3,
            const int* __restrict__ batch, float* __restrict__ aggB) {
    __shared__ HeadSh sh[WAVES];
    const int lane = threadIdx.x & 31;
    const int wid  = threadIdx.x >> 5;
    HeadSh& s = sh[wid];
    const int node = blockIdx.x * WAVES + wid;
    const bool valid = node < NL;

    for (int idx = lane; idx < 16 * H; idx += 32) {
        int t = idx >> 7, k = idx & (H - 1);
        float hv = valid ? hf[(long)node * H + k] : 0.0f;
        s.A[fragpos(t, k)] = (__bf16)(embed[t * H + k] * hv);
    }
    __syncthreads();

    const int mcol = lane & 15;
    const int half = lane >> 4;
    const v16bf* __restrict__ av1 = (const v16bf*)s.A;
    const v16bf* __restrict__ av2 = (const v16bf*)s.t1;
    const v16bf* __restrict__ bv1 = (const v16bf*)Wd1s;
    const v16bf* __restrict__ bv2 = (const v16bf*)Wd2s;

    // dense1: 128 -> 96 (85 live)
    for (int ct = 0; ct < 6; ++ct) {
        int n = (ct << 4) + mcol;
        v8f acc;
        float bcol = (n < 85) ? bd1[n] : 0.0f;
#pragma unroll
        for (int r = 0; r < 8; ++r) acc[r] = bcol;
#pragma unroll
        for (int kt = 0; kt < 4; ++kt) {
            v16bf a = av1[(kt << 5) + lane];
            v16bf b = bv1[(((ct << 2) + kt) << 5) + lane];
            acc = wmma_bf16(a, b, acc);
        }
#pragma unroll
        for (int r = 0; r < 8; ++r) {
            int row = r + (half << 3);
            s.t1[fragpos(row, n)] = (__bf16)silu_f(acc[r]);
        }
    }
    __syncthreads();

    // dense2: 96 -> 48 (43 live)
    for (int ct = 0; ct < 3; ++ct) {
        int n = (ct << 4) + mcol;
        v8f acc;
        float bcol = (n < 43) ? bd2[n] : 0.0f;
#pragma unroll
        for (int r = 0; r < 8; ++r) acc[r] = bcol;
#pragma unroll
        for (int kt = 0; kt < 3; ++kt) {
            v16bf a = av2[(kt << 5) + lane];
            v16bf b = bv2[((ct * 3 + kt) << 5) + lane];
            acc = wmma_bf16(a, b, acc);
        }
#pragma unroll
        for (int r = 0; r < 8; ++r) {
            int row = r + (half << 3);
            s.t2[row][n] = silu_f(acc[r]);
        }
    }
    __syncthreads();

    if (lane < 16) {
        float acc = bd3[0];
        for (int c = 0; c < 43; ++c) acc = fmaf(s.t2[lane][c], Wd3[c], acc);
        s.lg[lane] = acc;
    }
    __syncthreads();

    if (lane < 16 && valid) {
        float mx = -1e30f;
        for (int t = 0; t < 16; ++t) mx = fmaxf(mx, s.lg[t]);
        float se = 0.0f;
        for (int t = 0; t < 16; ++t) se += __expf(s.lg[t] - mx);
        float lgv = s.lg[lane] - mx - __logf(se);
        atomicAdd(&aggB[batch[node] * 16 + lane], lgv);
    }
}

__global__ void final_kernel(const float* __restrict__ aggB, float* __restrict__ out, int B) {
    int b = threadIdx.x;
    if (b < B) {
        float mx = -1e30f;
        for (int t = 0; t < 16; ++t) mx = fmaxf(mx, aggB[b * 16 + t]);
        float se = 0.0f;
        for (int t = 0; t < 16; ++t) se += __expf(aggB[b * 16 + t] - mx);
        float lse = mx + __logf(se);
        for (int t = 0; t < 16; ++t) out[b * 16 + t] = aggB[b * 16 + t] - lse;
    }
}

// ---------------------------------------------------------------- host driver
static inline int gdiv(long n, int b) { return (int)((n + b - 1) / b); }

extern "C" void kernel_launch(void* const* d_in, const int* in_sizes, int n_in,
                              void* d_out, int out_size, void* d_ws, size_t ws_size,
                              hipStream_t stream) {
    const float* ligand_x   = (const float*)d_in[0];
    const float* pocket_x   = (const float*)d_in[1];
    const float* ligand_pos = (const float*)d_in[2];
    const float* pocket_pos = (const float*)d_in[3];
    const float* Wl_emb = (const float*)d_in[4];
    const float* bl_emb = (const float*)d_in[5];
    const float* Wp_emb = (const float*)d_in[6];
    const float* bp_emb = (const float*)d_in[7];
    const float* We1 = (const float*)d_in[8];
    const float* be1 = (const float*)d_in[9];
    const float* We2 = (const float*)d_in[10];
    const float* be2 = (const float*)d_in[11];
    const float* Wx  = (const float*)d_in[12];
    const float* bx  = (const float*)d_in[13];
    const float* Wh1 = (const float*)d_in[14];
    const float* bh1 = (const float*)d_in[15];
    const float* Wh2 = (const float*)d_in[16];
    const float* bh2 = (const float*)d_in[17];
    const float* Wd1 = (const float*)d_in[18];
    const float* bd1 = (const float*)d_in[19];
    const float* Wd2 = (const float*)d_in[20];
    const float* bd2 = (const float*)d_in[21];
    const float* Wd3 = (const float*)d_in[22];
    const float* bd3 = (const float*)d_in[23];
    const int* l2l = (const int*)d_in[24];
    const int* p2p = (const int*)d_in[25];
    const int* lig_batch = (const int*)d_in[26];

    const int NL = in_sizes[0] / 16;
    const int NP = in_sizes[1] / 27;
    const int EL = in_sizes[24] / 2;
    const int EP = in_sizes[25] / 2;
    const int Bn = out_size / 16;

    char* p = (char*)d_ws;
    auto carve = [&](size_t bytes) -> void* {
        void* r = (void*)p;
        p += (bytes + 255) & ~(size_t)255;
        return r;
    };
    float*  hlf    = (float*)carve((size_t)NL * H * 4);
    float*  hpf    = (float*)carve((size_t)NP * H * 4);
    __bf16* hlb    = (__bf16*)carve((size_t)NL * H * 2);
    __bf16* hpb    = (__bf16*)carve((size_t)NP * H * 2);
    float*  xl     = (float*)carve((size_t)NL * 3 * 4);
    float*  xp2    = (float*)carve((size_t)NP * 3 * 4);
    float*  aggm_l = (float*)carve((size_t)NL * H * 4);
    float*  aggm_p = (float*)carve((size_t)NP * H * 4);
    float*  aggx_l = (float*)carve((size_t)NL * 3 * 4);
    float*  aggx_p = (float*)carve((size_t)NP * 3 * 4);
    float*  deg_l  = (float*)carve((size_t)NL * 4);
    float*  deg_p  = (float*)carve((size_t)NP * 4);
    __bf16* We1s   = (__bf16*)carve((size_t)6 * 256 * H * 2);
    __bf16* We2s   = (__bf16*)carve((size_t)6 * H * H * 2);
    __bf16* Wh1s   = (__bf16*)carve((size_t)6 * 256 * H * 2);
    __bf16* Wh2s   = (__bf16*)carve((size_t)6 * H * H * 2);
    __bf16* Wd1s   = (__bf16*)carve((size_t)128 * 96 * 2);
    __bf16* Wd2s   = (__bf16*)carve((size_t)96 * 48 * 2);
    float*  embed  = (float*)carve((size_t)16 * H * 4);
    float*  aggB   = (float*)carve((size_t)Bn * 16 * 4);

    // weight prep: fragment-swizzled bf16 copies
    for (int off = 0; off < 6; ++off) {
        swz_kernel<<<gdiv(256 * H, 256), 256, 0, stream>>>(
            We1 + (size_t)off * 257 * H, We1s + (size_t)off * 256 * H, 256, H, 256, H, H);
        swz_kernel<<<gdiv(H * H, 256), 256, 0, stream>>>(
            We2 + (size_t)off * H * H, We2s + (size_t)off * H * H, H, H, H, H, H);
        swz_kernel<<<gdiv(256 * H, 256), 256, 0, stream>>>(
            Wh1 + (size_t)off * 256 * H, Wh1s + (size_t)off * 256 * H, 256, H, 256, H, H);
        swz_kernel<<<gdiv(H * H, 256), 256, 0, stream>>>(
            Wh2 + (size_t)off * H * H, Wh2s + (size_t)off * H * H, H, H, H, H, H);
    }
    swz_kernel<<<gdiv(128 * 96, 256), 256, 0, stream>>>(Wd1, Wd1s, 128, 96, 128, 85, 85);
    swz_kernel<<<gdiv(96 * 48, 256), 256, 0, stream>>>(Wd2, Wd2s, 96, 48, 85, 43, 43);
    embed_type_kernel<<<gdiv(16 * H, 256), 256, 0, stream>>>(Wl_emb, bl_emb, embed);
    copy_kernel<<<gdiv((long)NL * 3, 256), 256, 0, stream>>>(ligand_pos, xl, (long)NL * 3);
    copy_kernel<<<gdiv((long)NP * 3, 256), 256, 0, stream>>>(pocket_pos, xp2, (long)NP * 3);

    embed_kernel<<<NL, H, 0, stream>>>(ligand_x, Wl_emb, bl_emb, hlf, hlb, NL, 16);
    embed_kernel<<<NP, H, 0, stream>>>(pocket_x, Wp_emb, bp_emb, hpf, hpb, NP, 27);

    for (int l = 0; l < 3; ++l) {
        for (int g = 0; g < 2; ++g) {
            const int E = g ? EP : EL;
            const int N = g ? NP : NL;
            const int* eidx = g ? p2p : l2l;
            float*  xg = g ? xp2 : xl;
            float*  hf = g ? hpf : hlf;
            __bf16* hb = g ? hpb : hlb;
            float*  am = g ? aggm_p : aggm_l;
            float*  ax = g ? aggx_p : aggx_l;
            float*  dg = g ? deg_p : deg_l;
            const int off = g * 3 + l;

            zero_kernel<<<gdiv((long)N * H, 256), 256, 0, stream>>>(am, (long)N * H);
            zero_kernel<<<gdiv((long)N * 3, 256), 256, 0, stream>>>(ax, (long)N * 3);
            zero_kernel<<<gdiv((long)N, 256), 256, 0, stream>>>(dg, (long)N);

            egcl_edge_kernel<<<gdiv((long)E, 16 * WAVES), 128, 0, stream>>>(
                eidx, E, xg, hb,
                We1s + (size_t)off * 256 * H, be1 + off * H,
                We1 + ((size_t)off * 257 + 256) * H,
                We2s + (size_t)off * H * H, be2 + off * H,
                Wx + off * H, bx + off,
                am, ax, dg);

            egcl_node_kernel<<<gdiv((long)N, 16 * WAVES), 128, 0, stream>>>(
                N, am, ax, dg,
                Wh1s + (size_t)off * 256 * H, bh1 + off * H,
                Wh2s + (size_t)off * H * H, bh2 + off * H,
                hf, hb, xg);
        }
    }

    zero_kernel<<<gdiv((long)Bn * 16, 256), 256, 0, stream>>>(aggB, (long)Bn * 16);
    head_kernel<<<gdiv((long)NL, WAVES), 128, 0, stream>>>(
        NL, hlf, embed, Wd1s, bd1, Wd2s, bd2, Wd3, bd3, lig_batch, aggB);
    final_kernel<<<1, 32, 0, stream>>>(aggB, (float*)d_out, Bn);
}
